// Style2ResidualBlock1D_54966991454475
// MI455X (gfx1250) — matrix operand order repaired and checked by reading, not compiled, not run on hardware
//
#include <hip/hip_runtime.h>
#include <cstdint>
#include <cstddef>

// Problem constants (match reference)
#define B_SZ   16
#define CIN    256
#define T_LEN  4096
#define COUT2  512
#define COUTH  256           // COUT2/2 (GLU)
#define KW     3
#define KD     (CIN * KW)    // 768 reduction dim
#define SDIM   128
#define LDSROW 40            // ushorts per LDS chunk row: 32 data + 8 pad (80B)

#if __has_builtin(__builtin_amdgcn_tensor_load_to_lds)
#define HAVE_TDM 1
#else
#define HAVE_TDM 0
#endif

typedef __attribute__((ext_vector_type(16))) __bf16 v16bf;
typedef __attribute__((ext_vector_type(8)))  float  v8f;
typedef __attribute__((ext_vector_type(4)))  unsigned int u32x4;
typedef __attribute__((ext_vector_type(2)))  unsigned int u32x2;
typedef __attribute__((ext_vector_type(4)))  float f32x4;
typedef __attribute__((ext_vector_type(8)))  int   i32x8;
typedef __attribute__((ext_vector_type(4)))  int   i32x4;

union FragBF { v16bf v; u32x4 q[2]; };

__device__ __forceinline__ unsigned short f2bf(float f) {
  // round-to-nearest-even fp32 -> bf16
  unsigned u = __float_as_uint(f);
  u += 0x7FFFu + ((u >> 16) & 1u);
  return (unsigned short)(u >> 16);
}

// ---------------------------------------------------------------------------
// Stage 1: s[b, cin] = (c_trg[b,:] . style_w[cin,:]) / sqrt(128) + style_b[cin]
// ---------------------------------------------------------------------------
__global__ __launch_bounds__(256) void style_linear_kernel(
    const float* __restrict__ c_trg, const float* __restrict__ style_w,
    const float* __restrict__ style_b, float* __restrict__ s_out)
{
  __shared__ float ct[SDIM];
  const int b = blockIdx.x;
  const int i = threadIdx.x;                 // cin, 0..255
  if (i < SDIM) ct[i] = c_trg[b * SDIM + i];
  __syncthreads();
  const float scale = 0.088388347648318447f; // 1/sqrt(128)
  const float* wrow = style_w + (size_t)i * SDIM;
  float acc = 0.f;
  #pragma unroll 8
  for (int j = 0; j < SDIM; ++j) acc = fmaf(ct[j], wrow[j], acc);
  s_out[b * CIN + i] = acc * scale + style_b[i];
}

// ---------------------------------------------------------------------------
// Stage 2: modulate + demodulate, pack bf16 A-matrix [B][COUT2][k*256+cin]
// ---------------------------------------------------------------------------
__global__ __launch_bounds__(64) void modulate_pack_kernel(
    const float* __restrict__ base_w, const float* __restrict__ s,
    unsigned short* __restrict__ packA)
{
  const int cout = blockIdx.x;
  const int b    = blockIdx.y;
  const int tid  = threadIdx.x;
  __shared__ float red[64];
  const float conv_scale = 1.0f / 48.0f;     // 1/sqrt(256*3^2)
  const float* bw = base_w + (size_t)cout * CIN * KW;
  const float* sb = s + b * CIN;

  float acc = 0.f;
  for (int ci = tid; ci < CIN; ci += 64) {
    const float sv = sb[ci] * conv_scale;
    const float w0 = bw[ci*3+0]*sv, w1 = bw[ci*3+1]*sv, w2 = bw[ci*3+2]*sv;
    acc += w0*w0 + w1*w1 + w2*w2;
  }
  red[tid] = acc;
  __syncthreads();
  #pragma unroll
  for (int off = 32; off > 0; off >>= 1) {
    if (tid < off) red[tid] += red[tid + off];
    __syncthreads();
  }
  const float demod = rsqrtf(red[0] + 1e-8f);

  unsigned short* dst = packA + ((size_t)b * COUT2 + cout) * KD;
  for (int ci = tid; ci < CIN; ci += 64) {
    const float sv = sb[ci] * conv_scale * demod;
    dst[0*CIN + ci] = f2bf(bw[ci*3+0]*sv);
    dst[1*CIN + ci] = f2bf(bw[ci*3+1]*sv);
    dst[2*CIN + ci] = f2bf(bw[ci*3+2]*sv);
  }
}

// ---------------------------------------------------------------------------
// TDM descriptor: DMA one weight tile (32 cin x 3 taps x 16 cout rows, bf16)
// from global into LDS, with hardware padding 64B data + 16B pad per chunk
// (pad_interval=16 DWORDs -> code 3; pad_amount=4 DWORDs -> code 3), matching
// the 80-byte LDSROW layout.  D# field packing per CDNA5 ISA.
// This toolchain's builtin is the 6-arg clang-23 form:
//   (uint32x4 g0, int32x8 g1, int32x4 g2, int32x4 g3, int32x8 extra, i32 cpol)
// The extra int32x8 operand is uncharacterized -> pass reserved zeros.
// ---------------------------------------------------------------------------
#if HAVE_TDM
__device__ __forceinline__ void tdm_load_A(unsigned lds_addr, const unsigned short* gptr) {
  const unsigned long long ga = (unsigned long long)(uintptr_t)gptr;
  u32x4 g0;
  g0[0] = 1u;                                              // count=1, user mode, load
  g0[1] = lds_addr;                                        // lds_addr [63:32]
  g0[2] = (unsigned)(ga & 0xffffffffu);                    // global_addr lo
  g0[3] = (unsigned)((ga >> 32) & 0x01ffffffu) | (2u << 30); // addr[56:32] | type=2
  i32x8 g1;
  g1[0] = (int)((1u << 16) | (1u << 20) | (3u << 22) | (3u << 25));
         // data_size=2B | pad_enable | pad_interval(16 DW) | pad_amount(4 DW)
  g1[1] = (int)(32u << 16);          // tensor_dim0 = 32 (lo16 here)
  g1[2] = (int)(3u << 16);           // tensor_dim0 hi=0, tensor_dim1 = 3
  g1[3] = (int)(32u << 16);          // tensor_dim1 hi=0, tile_dim0 = 32
  g1[4] = (int)(3u | (16u << 16));   // tile_dim1 = 3 taps, tile_dim2 = 16 rows
  g1[5] = 256;                       // tensor_dim0_stride = 256 elems (tap stride)
  g1[6] = (int)(768u << 16);         // tensor_dim1_stride = 768 elems (row stride)
  g1[7] = 0;
  i32x4 g2;
  g2[0] = 16;                        // tensor_dim2 = 16
  g2[1] = 0;                         // tensor_dim3 unused
  g2[2] = 768;                       // tensor_dim2_stride = 768
  g2[3] = 0;                         // tile_dim3 = 0 (unused)
  i32x4 g3 = {0, 0, 0, 0};
  i32x8 gx = {0, 0, 0, 0, 0, 0, 0, 0};  // uncharacterized extra operand: zeros
  __builtin_amdgcn_tensor_load_to_lds(g0, g1, g2, g3, gx, 0);
}
#endif

// ---------------------------------------------------------------------------
// Stage 3: per-sample conv1d as WMMA GEMM + fused GLU.
// Block = 128 threads (4 wave32). Block tile: 16 couts (both GLU halves) x 64 t.
// Weights: TDM double-buffered into LDS (compute cs&1 while DMA fills (cs+1)&1,
// synchronized on TENSORcnt).  x: VALU fp32->bf16 convert + transpose into LDS.
// ---------------------------------------------------------------------------
__global__ __launch_bounds__(128) void conv_glu_wmma_kernel(
    const float* __restrict__ x, const unsigned short* __restrict__ packA,
    float* __restrict__ out)
{
  // ldsA layout: [buf(2)][part(2)][row(16)][tap(3)][LDSROW] (TDM write order)
  __shared__ __align__(16) unsigned short ldsA[2 * 2 * 16 * 3 * LDSROW]; // 15360 B
  __shared__ __align__(16) unsigned short ldsX[80 * LDSROW];             // 6400 B
  #define APART_US (16 * 3 * LDSROW)      // one GLU-half tile, in ushorts
  #define ABUF_US  (2 * APART_US)         // one double-buffer slot, in ushorts

  const int m0 = blockIdx.x * 16;   // cout tile within GLU half
  const int t0 = blockIdx.y * 64;   // time tile
  const int b  = blockIdx.z;
  const int tid  = threadIdx.x;
  const int lane = tid & 31;
  const int wave = tid >> 5;
  const int row  = lane & 15;       // M row (A/C) / N col (B)
  const int half = lane >> 4;       // lane-group: K-split (A/B), M+8 (C)

  const float* xb = x + (size_t)b * CIN * T_LEN;
  const unsigned short* Ab = packA + (size_t)b * COUT2 * KD;

  v8f acc_a = {0.f,0.f,0.f,0.f,0.f,0.f,0.f,0.f};
  v8f acc_b = {0.f,0.f,0.f,0.f,0.f,0.f,0.f,0.f};

#if HAVE_TDM
  const unsigned ldsA_base = (unsigned)(uintptr_t)&ldsA[0];
  if (wave == 0) {  // prologue: DMA cs=0 weight tiles into buffer 0
    tdm_load_A(ldsA_base,                Ab + (size_t)m0 * KD);
    tdm_load_A(ldsA_base + APART_US * 2, Ab + (size_t)(m0 + COUTH) * KD);
  }
#endif

  for (int cs = 0; cs < 8; ++cs) {
    const int cin0 = cs * 32;
    const int cbuf = cs & 1;

#if HAVE_TDM
    if (wave == 0 && cs < 7) {  // DMA next channel-group's weights into other buffer
      const int nb = (cs + 1) & 1;
      tdm_load_A(ldsA_base + (unsigned)(nb * ABUF_US) * 2u,
                 Ab + (size_t)m0 * KD + (cin0 + 32));
      tdm_load_A(ldsA_base + (unsigned)(nb * ABUF_US + APART_US) * 2u,
                 Ab + (size_t)(m0 + COUTH) * KD + (cin0 + 32));
    }
#else
    // fallback: VALU staging of the current weight tile into buffer cs&1
    for (int s = tid; s < 384; s += 128) {
      const int q    = s & 3;
      const int t1   = s >> 2;
      const int kk3  = t1 % 3;
      const int t2   = t1 / 3;
      const int r    = t2 & 15;
      const int part = t2 >> 4;
      const u32x4 v = *(const u32x4*)(Ab + (size_t)(m0 + part * COUTH + r) * KD
                                      + kk3 * CIN + cin0 + q * 8);
      *(u32x4*)(&ldsA[(((cbuf * 2 + part) * 16 + r) * 3 + kk3) * LDSROW + q * 8]) = v;
    }
#endif

    if (cs < 6) { // warm GL2 for the DMA two tiles ahead (global_prefetch_b8)
      __builtin_prefetch(Ab + (size_t)(m0 + (tid & 15)) * KD + cin0 + 64, 0, 3);
    }

    // --- stage X tile: channels [cin0,cin0+32), times [t0-8, t0+72), bf16 [t][c] ---
    for (int s = tid; s < 160; s += 128) {
      const int tg4 = s >> 3;             // time quad 0..19
      const int c4  = (s & 7) * 4;        // channel group of 4
      const int tg  = t0 - 8 + tg4 * 4;   // global start time (16B aligned)
      float v[4][4];
      if (tg >= 0 && tg + 4 <= T_LEN) {
        #pragma unroll
        for (int cc = 0; cc < 4; ++cc) {
          const f32x4 f = *(const f32x4*)(xb + (size_t)(cin0 + c4 + cc) * T_LEN + tg);
          v[cc][0]=f[0]; v[cc][1]=f[1]; v[cc][2]=f[2]; v[cc][3]=f[3];
        }
      } else { // conv halo: zero-pad out-of-range t
        #pragma unroll
        for (int cc = 0; cc < 4; ++cc)
          #pragma unroll
          for (int tt = 0; tt < 4; ++tt) {
            const int t = tg + tt;
            v[cc][tt] = (t >= 0 && t < T_LEN) ? xb[(size_t)(cin0 + c4 + cc) * T_LEN + t] : 0.f;
          }
      }
      #pragma unroll
      for (int tt = 0; tt < 4; ++tt) {    // transpose-pack: 4 bf16 channels per b64 store
        u32x2 p;
        p[0] = (unsigned)f2bf(v[0][tt]) | ((unsigned)f2bf(v[1][tt]) << 16);
        p[1] = (unsigned)f2bf(v[2][tt]) | ((unsigned)f2bf(v[3][tt]) << 16);
        *(u32x2*)(&ldsX[(tg4 * 4 + tt) * LDSROW + c4]) = p;
      }
    }

#if HAVE_TDM
    if (wave == 0) {  // current buffer's pair must have landed; next pair may fly
      if (cs < 7) __builtin_amdgcn_s_wait_tensorcnt(2);
      else        __builtin_amdgcn_s_wait_tensorcnt(0);
    }
#endif
    __syncthreads();

    // --- 3 taps: X tile reused, shift by +k on the LDS time row ---
    #pragma unroll
    for (int k = 0; k < KW; ++k) {
      FragBF fa, fb, fx;
      // A frag (16x32): lane row=M; K = 8*half + l (l<8), 16 + 8*half + (l-8) (l>=8)
      const unsigned short* pa =
          &ldsA[(((cbuf * 2 + 0) * 16 + row) * 3 + k) * LDSROW + half * 8];
      fa.q[0] = *(const u32x4*)pa;
      fa.q[1] = *(const u32x4*)(pa + 16);
      const unsigned short* pb =
          &ldsA[(((cbuf * 2 + 1) * 16 + row) * 3 + k) * LDSROW + half * 8];
      fb.q[0] = *(const u32x4*)pb;
      fb.q[1] = *(const u32x4*)(pb + 16);
      // B frag (32x16): lane col N=row (time), K = 8*half + l (l<8), 16 + 8*half + (l-8) (l>=8) (channels)
      const unsigned short* px = &ldsX[(wave * 16 + row + 7 + k) * LDSROW + half * 8];
      fx.q[0] = *(const u32x4*)px;
      fx.q[1] = *(const u32x4*)(px + 16);

      acc_a = __builtin_amdgcn_wmma_f32_16x16x32_bf16(false, fa.v, false, fx.v,
                                                      (short)0, acc_a, false, false);
      acc_b = __builtin_amdgcn_wmma_f32_16x16x32_bf16(false, fb.v, false, fx.v,
                                                      (short)0, acc_b, false, false);
    }
    __syncthreads();
  }

  // --- fused GLU epilogue: out = a * sigmoid(b) ---
  float* ob = out + (size_t)b * COUTH * T_LEN;
  const int tcol = t0 + wave * 16 + row;   // N = lane&15
  #pragma unroll
  for (int r = 0; r < 8; ++r) {            // C layout: M = r + 8*half
    const int cout = m0 + r + 8 * half;
    const float av = acc_a[r];
    const float bv = acc_b[r];
    ob[(size_t)cout * T_LEN + tcol] = av / (1.0f + __expf(-bv));
  }
}

// ---------------------------------------------------------------------------
extern "C" void kernel_launch(void* const* d_in, const int* in_sizes, int n_in,
                              void* d_out, int out_size, void* d_ws, size_t ws_size,
                              hipStream_t stream) {
  const float* x       = (const float*)d_in[0];
  // d_in[1] = c_src (unused by reference path)
  const float* c_trg   = (const float*)d_in[2];
  const float* style_w = (const float*)d_in[3];
  const float* style_b = (const float*)d_in[4];
  const float* base_w  = (const float*)d_in[5];
  float* out = (float*)d_out;

  // workspace: s [16,256] f32 (16 KB) + packed bf16 weights [16,512,768] (~12.6 MB)
  float* s_buf = (float*)d_ws;
  unsigned short* packA = (unsigned short*)((char*)d_ws + 16384);

  style_linear_kernel<<<dim3(B_SZ), dim3(CIN), 0, stream>>>(c_trg, style_w, style_b, s_buf);
  modulate_pack_kernel<<<dim3(COUT2, B_SZ), dim3(64), 0, stream>>>(base_w, s_buf, packA);
  conv_glu_wmma_kernel<<<dim3(COUTH / 16, T_LEN / 64, B_SZ), dim3(128), 0, stream>>>(x, packA, out);

  (void)in_sizes; (void)n_in; (void)out_size; (void)ws_size;
}
